// CompactBilinearPooling_3066606649803
// MI455X (gfx1250) — matrix operand order repaired and checked by reading, not compiled, not run on hardware
//
#include <hip/hip_runtime.h>
#include <math.h>

// Compact Bilinear Pooling via Gram-matrix + bucket scatter (exact equivalent
// of count-sketch/FFT circular convolution), WMMA f16->f32 on gfx1250.
// Staging uses GLOBAL_LOAD_ASYNC_TO_LDS_B128 (ASYNCcnt) when available.

typedef __attribute__((ext_vector_type(16))) _Float16 v16h;
typedef __attribute__((ext_vector_type(8)))  _Float16 v8h;
typedef __attribute__((ext_vector_type(8)))  float    v8f;
typedef __attribute__((ext_vector_type(4)))  int      v4i;

#if defined(__has_builtin)
#  if __has_builtin(__builtin_amdgcn_global_load_async_to_lds_b128)
#    define USE_ASYNC_LDS 1
#  endif
#endif
#ifndef USE_ASYNC_LDS
#  define USE_ASYNC_LDS 0
#endif

#define B_    16
#define C_    2048
#define HW_   196
#define KPAD  224          // 7 * 32, zero-padded K
#define D_    8192         // power of two -> mod is AND
#define TILE  128          // Gram tile: 128x128 per block
#define KC    32           // K chunk per WMMA step
#define NKC   (KPAD / KC)  // 7

static __device__ __forceinline__ void wait_async_lds() {
#if defined(__has_builtin)
#  if __has_builtin(__builtin_amdgcn_s_wait_asynccnt)
  __builtin_amdgcn_s_wait_asynccnt(0);
  return;
#  endif
#endif
  asm volatile("s_wait_asynccnt 0x0" ::: "memory");
}

// ---------------------------------------------------------------------------
// Prep: xs1[b,c,k] = f16( x[b,c,k] * (2*s1[c]-1) ), xs2 likewise, K padded.
// ---------------------------------------------------------------------------
__global__ void cbp_prep(const float* __restrict__ x,
                         const int* __restrict__ s1, const int* __restrict__ s2,
                         _Float16* __restrict__ xs1, _Float16* __restrict__ xs2) {
  int idx = blockIdx.x * blockDim.x + threadIdx.x;
  const int total = B_ * C_ * KPAD;
  if (idx >= total) return;
  int k  = idx % KPAD;
  int bc = idx / KPAD;
  int c  = bc % C_;
  float v = (k < HW_) ? x[(size_t)bc * HW_ + k] : 0.0f;
  float g1 = (float)(2 * s1[c] - 1);
  float g2 = (float)(2 * s2[c] - 1);
  xs1[idx] = (_Float16)(v * g1);
  xs2[idx] = (_Float16)(v * g2);
}

__global__ void cbp_zero(float* __restrict__ p) {
  p[blockIdx.x * blockDim.x + threadIdx.x] = 0.0f;
}

// ---------------------------------------------------------------------------
// Gram tile (128x128) via v_wmma_f32_16x16x32_f16, scatter into LDS histogram
// of all D buckets, then flush with global f32 atomics (accumulator is L2
// resident: 512 KB).
// Block = 256 threads = 8 waves; wave w owns row-block w (16 rows), all 8
// column blocks.  grid = (C/TILE, C/TILE, B).
// ---------------------------------------------------------------------------
__global__ __launch_bounds__(256)
void cbp_gram(const _Float16* __restrict__ xs1, const _Float16* __restrict__ xs2,
              const int* __restrict__ h1, const int* __restrict__ h2,
              float* __restrict__ acc_g) {
  __shared__ __align__(32) _Float16 shA[TILE * KC];   // 8 KB
  __shared__ __align__(32) _Float16 shB[TILE * KC];   // 8 KB
  __shared__ int   shH[2 * TILE];                     // 1 KB
  __shared__ float hist[D_];                          // 32 KB

  const int ti = blockIdx.x, tj = blockIdx.y, b = blockIdx.z;
  const int t    = threadIdx.x;
  const int lane = t & 31;
  const int wv   = t >> 5;

  // hash buckets for this tile's rows / cols
  if (t < TILE) shH[t] = h1[ti * TILE + t];
  else          shH[t] = h2[tj * TILE + (t - TILE)];
  // zero the bucket histogram (32 floats per thread)
  for (int i = t; i < D_; i += 256) hist[i] = 0.0f;

  const _Float16* Abase = xs1 + ((size_t)b * C_ + (size_t)ti * TILE) * KPAD;
  const _Float16* Bbase = xs2 + ((size_t)b * C_ + (size_t)tj * TILE) * KPAD;

  v8f acc[8];
  for (int i = 0; i < 8; ++i) acc[i] = {};

  for (int kc = 0; kc < NKC; ++kc) {
    __syncthreads();  // protect shA/shB against previous iteration readers
    // Stage one 128x32 chunk of A and B: 4096 halves each; 16 halves/thread
    // as 2x 16B transfers per matrix.
    for (int p = 0; p < 2; ++p) {
      int hoff = t * 16 + p * 8;      // 0..4095, 8-half (16B) granules
      int row  = hoff >> 5;
      int kk   = hoff & 31;
      const _Float16* ga = Abase + (size_t)row * KPAD + kc * KC + kk;
      const _Float16* gb = Bbase + (size_t)row * KPAD + kc * KC + kk;
#if USE_ASYNC_LDS
      // memory -> LDS without VGPR roundtrip; tracked by ASYNCcnt
      __builtin_amdgcn_global_load_async_to_lds_b128(
          (v4i*)(void*)ga, (v4i*)(void*)(shA + hoff), 0, 0);
      __builtin_amdgcn_global_load_async_to_lds_b128(
          (v4i*)(void*)gb, (v4i*)(void*)(shB + hoff), 0, 0);
#else
      *(v8h*)(shA + hoff) = *(const v8h*)ga;
      *(v8h*)(shB + hoff) = *(const v8h*)gb;
#endif
    }
    if (kc + 1 < NKC) {  // prefetch next K chunk -> global_prefetch_b8
      __builtin_prefetch(Abase + (size_t)(t >> 1) * KPAD + (kc + 1) * KC, 0, 1);
      __builtin_prefetch(Bbase + (size_t)(t >> 1) * KPAD + (kc + 1) * KC, 0, 1);
    }
#if USE_ASYNC_LDS
    wait_async_lds();   // drain this wave's async LDS fills
#endif
    __syncthreads();    // all waves' fills visible

    // A fragment (16x32 f16, ISA wave32 layout): lane<16 -> M=lane, K runs
    // {0..7,16..23}; lane>=16 -> M=lane-16, K runs {8..15,24..31}.
    const int arow  = wv * 16 + (lane & 15);
    const int kbase = (lane >= 16) ? 8 : 0;
    v8h alo = *(const v8h*)(shA + arow * KC + kbase);
    v8h ahi = *(const v8h*)(shA + arow * KC + kbase + 16);
    v16h afrag = __builtin_shufflevector(alo, ahi,
        0, 1, 2, 3, 4, 5, 6, 7, 8, 9, 10, 11, 12, 13, 14, 15);

    for (int nb = 0; nb < 8; ++nb) {
      // B fragment (32x16 f16): lane<16 -> col N=lane, K runs {0..7,16..23};
      // lane>=16 -> N=lane-16, K runs {8..15,24..31} (matches A).
      const int brow = nb * 16 + (lane & 15);
      const int kb2  = (lane >= 16) ? 8 : 0;
      v8h blo = *(const v8h*)(shB + brow * KC + kb2);
      v8h bhi = *(const v8h*)(shB + brow * KC + kb2 + 16);
      v16h bfrag = __builtin_shufflevector(blo, bhi,
          0, 1, 2, 3, 4, 5, 6, 7, 8, 9, 10, 11, 12, 13, 14, 15);
      acc[nb] = __builtin_amdgcn_wmma_f32_16x16x32_f16(
          false, afrag, false, bfrag, (short)0, acc[nb], false, false);
    }
  }

  // Scatter: D-matrix layout -> VGPR vg holds M = vg + (lane>=16 ? 8 : 0),
  // N = lane & 15.  bucket = (h1[i] + h2[j]) & (D-1).
  const int mbase = wv * 16 + ((lane >= 16) ? 8 : 0);
  const int ncol  = lane & 15;
  for (int nb = 0; nb < 8; ++nb) {
    const int jh = shH[TILE + nb * 16 + ncol];
    for (int vg = 0; vg < 8; ++vg) {
      const int bucket = (shH[mbase + vg] + jh) & (D_ - 1);
      atomicAdd(&hist[bucket], acc[nb][vg]);   // ds_add_f32
    }
  }
  __syncthreads();

  // Flush histogram to the global per-batch accumulator (L2 resident).
  float* accb = acc_g + (size_t)b * D_;
  for (int i = t; i < D_; i += 256) {
    float v = hist[i];
    if (v != 0.0f) atomicAdd(accb + i, v);     // global_atomic_add_f32
  }
}

// ---------------------------------------------------------------------------
// Finalize: signed sqrt + L2 normalize per batch.  One block per batch.
// ---------------------------------------------------------------------------
__global__ __launch_bounds__(256)
void cbp_finalize(const float* __restrict__ acc_g, float* __restrict__ out) {
  __shared__ float red[256];
  const int b = blockIdx.x, t = threadIdx.x;
  const float* a = acc_g + (size_t)b * D_;
  float ss[D_ / 256];   // 32 values per thread, stays in VGPRs
  float psum = 0.0f;
  for (int i = 0; i < D_ / 256; ++i) {
    float v = a[t + i * 256];
    float s = copysignf(sqrtf(fabsf(v) + 1e-5f), v);
    ss[i] = s;
    psum += s * s;
  }
  red[t] = psum;
  __syncthreads();
  for (int off = 128; off > 0; off >>= 1) {
    if (t < off) red[t] += red[t + off];
    __syncthreads();
  }
  const float norm = fmaxf(sqrtf(red[0]), 1e-12f);
  for (int i = 0; i < D_ / 256; ++i)
    out[(size_t)b * D_ + t + i * 256] = ss[i] / norm;
}

// ---------------------------------------------------------------------------
extern "C" void kernel_launch(void* const* d_in, const int* in_sizes, int n_in,
                              void* d_out, int out_size, void* d_ws, size_t ws_size,
                              hipStream_t stream) {
  const float* x  = (const float*)d_in[0];
  const int*   h1 = (const int*)d_in[1];
  const int*   h2 = (const int*)d_in[2];
  const int*   s1 = (const int*)d_in[3];
  const int*   s2 = (const int*)d_in[4];
  float* out = (float*)d_out;

  // Workspace carve-up (~30 MB): xs1 | xs2 (f16, K padded) | acc (f32 B*D)
  char* ws = (char*)d_ws;
  _Float16* xs1 = (_Float16*)ws;
  _Float16* xs2 = xs1 + (size_t)B_ * C_ * KPAD;
  float*    acc = (float*)(xs2 + (size_t)B_ * C_ * KPAD);

  const int totalPrep = B_ * C_ * KPAD;                 // 7,340,032
  cbp_prep<<<(totalPrep + 255) / 256, 256, 0, stream>>>(x, s1, s2, xs1, xs2);
  cbp_zero<<<(B_ * D_) / 256, 256, 0, stream>>>(acc);

  dim3 grid(C_ / TILE, C_ / TILE, B_);                  // 16 x 16 x 16
  cbp_gram<<<grid, 256, 0, stream>>>(xs1, xs2, h1, h2, acc);

  cbp_finalize<<<B_, 256, 0, stream>>>(acc, out);
}
